// CapsLayer_47725676593516
// MI455X (gfx1250) — matrix operand order repaired and checked by reading, not compiled, not run on hardware
//
#include <hip/hip_runtime.h>
#include <hip/hip_bf16.h>
#include <stdint.h>

// Problem dims (from reference): B=128, I=512, D=256, J=32, V=32
#define Bc 128
#define Ic 512
#define Dc 256
#define Jc 32
#define Vc 32
#define PADW 264   // 256 + 8 halves: row stride 528B = 33*16B (16B aligned, bank-advance 4/row)

typedef __attribute__((ext_vector_type(16))) _Float16 v16h;
typedef __attribute__((ext_vector_type(8)))  _Float16 v8h;
typedef __attribute__((ext_vector_type(8)))  float    v8f;
typedef __attribute__((ext_vector_type(4)))  uint32_t v4u;
typedef __attribute__((ext_vector_type(8)))  uint32_t v8u;

// ---------------------------------------------------------------------------
// TDM: DMA one contiguous run of `nElems` f32 from global into LDS.
// D# packed per CDNA5 ISA §8: group0 = {ctrl, lds_addr, gaddr_lo, gaddr_hi|type},
// group1 = dims/strides (data_size=4B, 1 row of nElems), groups 2/3 zeroed
// (tile_dim2..4 = 0 -> unused). Issued by one wave only (EXEC is ignored by
// TDM, so every issuing wave would start its own DMA). Completion tracked with
// TENSORcnt.
// ---------------------------------------------------------------------------
__device__ inline void tdm_load_f32(const float* gsrc, uint32_t ldsByteAddr, int nElems)
{
    uint64_t ga = (uint64_t)(uintptr_t)gsrc;
    v4u g0;
    g0[0] = 1u;                                   // count=1 (valid), user mode
    g0[1] = ldsByteAddr;                          // LDS destination (bytes)
    g0[2] = (uint32_t)ga;                         // global_addr[31:0]
    g0[3] = (uint32_t)((ga >> 32) & 0x01FFFFFFu)  // global_addr[56:32]
          | 0x80000000u;                          // type=2 ("image")
    v8u g1;
    g1[0] = 0x00020000u;                          // wg_mask=0, data_size=2 (4B)
    g1[1] = ((uint32_t)nElems & 0xFFFFu) << 16;   // tensor_dim0[15:0]
    g1[2] = (((uint32_t)nElems >> 16) & 0xFFFFu)  // tensor_dim0[31:16]
          | (1u << 16);                           // tensor_dim1 = 1 (lo16)
    g1[3] = ((uint32_t)nElems) << 16;             // tile_dim0 = nElems (<=65535)
    g1[4] = 1u;                                   // tile_dim1 = 1, tile_dim2 = 0
    g1[5] = (uint32_t)nElems;                     // tensor_dim0_stride[31:0]
    g1[6] = 0u;                                   // stride hi / dim1_stride lo
    g1[7] = 0u;
    v4u gz = {0u, 0u, 0u, 0u};
    asm volatile("tensor_load_to_lds %0, %1, %2, %3"
                 :
                 : "s"(g0), "s"(g1), "s"(gz), "s"(gz)
                 : "memory");
}

// ---------------------------------------------------------------------------
// Kernel 1: u_hat[b,i,j,v] = sum_d u[b,i,d] * W[i,j,d,v]
// One workgroup per i. 256 threads = 8 waves; wave w owns rows b in [16w,16w+16).
// A (u slice, 128x256) staged once in LDS as f16; 8 A-fragments hoisted to
// registers and reused across all 32 j. Each W tile (32KB, contiguous) is
// DMA'd by the Tensor Data Mover into an f32 staging buffer (in flight under
// the previous j's WMMAs), then converted f32->f16 transposed [v][d] into the
// double-buffered WMMA tile. 8 K-steps of v_wmma_f32_16x16x32_f16 per j.
// ---------------------------------------------------------------------------
__global__ __launch_bounds__(256, 1)
void caps_uhat_gemm(const float* __restrict__ x,   // [B,I,D,1]
                    const float* __restrict__ W,   // [1,I,J,D,V]
                    float* __restrict__ uhat)      // [B,I,J,V]
{
    const int i    = blockIdx.x;
    const int t    = threadIdx.x;
    const int lane = t & 31;
    const int wave = t >> 5;
    const int half = lane >> 4;      // 0 = lanes 0-15, 1 = lanes 16-31
    const int l16  = lane & 15;

    __shared__ _Float16 uA[Bc * PADW];        // 67,584 B
    __shared__ _Float16 wB[2][Vc * PADW];     // 33,792 B  (transposed tile [v][d])
    __shared__ float    wF32[Dc * Vc];        // 32,768 B  (raw TDM landing buffer)

    // ---- stage u[:, i, :] -> uA (f32 -> f16) ----
    {
        const float* up = x + (size_t)i * Dc;     // + b*I*D + d
        #pragma unroll 4
        for (int p = 0; p < 128; ++p) {
            int idx = t + p * 256;                // 0 .. 32767
            int b = idx >> 8;
            int d = idx & 255;
            uA[b * PADW + d] = (_Float16)up[(size_t)b * (Ic * Dc) + d];
        }
    }

    const float*   wbase    = W + (size_t)i * (Jc * Dc * Vc);
    const uint32_t wf32_lds = (uint32_t)(uintptr_t)(const void*)wF32;

    // LDS->LDS convert: raw f32 [d][v] tile -> f16 transposed [v][d]
    auto convertW = [&](int buf) {
        #pragma unroll 4
        for (int p = 0; p < 32; ++p) {
            int idx = t + p * 256;                // 0 .. 8191
            int d = idx >> 5;
            int v = idx & 31;
            wB[buf][v * PADW + d] = (_Float16)wF32[idx];
        }
    };

    // ---- prologue: TDM tile j=0, convert, launch TDM tile j=1 ----
    if (wave == 0) {
        tdm_load_f32(wbase, wf32_lds, Dc * Vc);
        __builtin_amdgcn_s_wait_tensorcnt(0);
    }
    __syncthreads();                 // TDM done + uA stores visible
    convertW(0);
    __syncthreads();                 // wF32 free for reuse
    if (wave == 0) tdm_load_f32(wbase + (Dc * Vc), wf32_lds, Dc * Vc);

    // ---- hoist A fragments (reused for every j) ----
    // ISA 16-bit A 16x32 layout: lanes 0-15 hold K = k0+{0..7, 16..23},
    // lanes 16-31 hold K = k0+{8..15, 24..31}; row = m0 + (lane&15).
    const int row = wave * 16 + l16;
    v16h afrag[8];
    #pragma unroll
    for (int kc = 0; kc < 8; ++kc) {
        const _Float16* pa = &uA[row * PADW + kc * 32 + half * 8];
        v8h lo = *(const v8h*)pa;          // K = k0+8h .. k0+8h+7
        v8h hi = *(const v8h*)(pa + 16);   // K = k0+16+8h .. k0+16+8h+7
        v16h a;
        #pragma unroll
        for (int e = 0; e < 8; ++e) { a[e] = lo[e]; a[8 + e] = hi[e]; }
        afrag[kc] = a;
    }

    // ---- j loop: TDM(j+2) in flight under compute of j+1 ----
    for (int j = 0; j < Jc; ++j) {
        const int buf = j & 1;

        v8f acc0 = {};   // cols 0..15
        v8f acc1 = {};   // cols 16..31
        #pragma unroll
        for (int kc = 0; kc < 8; ++kc) {
            // ISA 16-bit B KxN layout: lanes 0-15 hold K=k0+{0..7, 16..23} of col=lane,
            // lanes 16-31 hold K=k0+{8..15, 24..31}. Tile stored as [v][d] so each
            // 8-run of K-values per lane is contiguous.
            const _Float16* pb0 = &wB[buf][(l16)      * PADW + kc * 32 + half * 8];
            const _Float16* pb1 = &wB[buf][(l16 + 16) * PADW + kc * 32 + half * 8];
            v8h b0lo = *(const v8h*)pb0;
            v8h b0hi = *(const v8h*)(pb0 + 16);
            v8h b1lo = *(const v8h*)pb1;
            v8h b1hi = *(const v8h*)(pb1 + 16);
            v16h bf0, bf1;
            #pragma unroll
            for (int e = 0; e < 8; ++e) {
                bf0[e] = b0lo[e]; bf0[8 + e] = b0hi[e];
                bf1[e] = b1lo[e]; bf1[8 + e] = b1hi[e];
            }
            acc0 = __builtin_amdgcn_wmma_f32_16x16x32_f16(
                       false, afrag[kc], false, bf0, (short)0, acc0, false, false);
            acc1 = __builtin_amdgcn_wmma_f32_16x16x32_f16(
                       false, afrag[kc], false, bf1, (short)0, acc1, false, false);
        }

        // ---- writeback: C layout — VGPR g: row = m0 + 8*half + g, col = lane&15 (+16) ----
        #pragma unroll
        for (int g = 0; g < 8; ++g) {
            int bb = wave * 16 + half * 8 + g;
            int o  = ((bb * Ic + i) * Jc + j) * Vc;    // < 2^26, int is safe
            uhat[o + l16]      = acc0[g];
            uhat[o + 16 + l16] = acc1[g];
        }

        // ---- stage tile j+1 into wB[buf^1]; kick TDM for tile j+2 ----
        if (j + 1 < Jc) {
            if (wave == 0) __builtin_amdgcn_s_wait_tensorcnt(0);
            __syncthreads();                  // TDM(j+1) landed; wB[buf^1] readers done
            convertW(buf ^ 1);
            __syncthreads();                  // wF32 free for reuse
            if (j + 2 < Jc && wave == 0)
                tdm_load_f32(wbase + (size_t)(j + 2) * (Dc * Vc), wf32_lds, Dc * Vc);
        }
    }
}

// ---------------------------------------------------------------------------
// Kernel 2: dynamic routing (3 iterations) + squash, one workgroup per b.
// 1024 threads = 32 waves; wave = j, lane = v. b_IJ kept per-lane in registers
// (lane v owns i = v + 32k); c_IJ row-private in LDS for broadcast reads.
// u_hat is 256MB; each workgroup rereads only its own 2MB b-slice in all three passes.
// ---------------------------------------------------------------------------
__global__ __launch_bounds__(1024, 1)
void caps_routing(const float* __restrict__ uhat,  // [B,I,J,V]
                  float* __restrict__ out)         // [B,J,V,1]
{
    const int b = blockIdx.x;
    const int j = threadIdx.x >> 5;
    const int v = threadIdx.x & 31;

    __shared__ float cI[Jc][Ic];   // 64 KB, row j private to wave j

    float breg[16];                // b_IJ[i= v+32k][j]
    #pragma unroll
    for (int k = 0; k < 16; ++k) breg[k] = 0.0f;

    const float* up = uhat + ((size_t)b * Ic * Jc + j) * Vc + v;  // index i -> up[i*1024]
    float vjv = 0.0f;

    for (int r = 0; r < 3; ++r) {
        // ---- c[:,j] = I * softmax_i(b[:,j]) ----
        float m = breg[0];
        #pragma unroll
        for (int k = 1; k < 16; ++k) m = fmaxf(m, breg[k]);
        #pragma unroll
        for (int o = 16; o > 0; o >>= 1) m = fmaxf(m, __shfl_xor(m, o, 32));

        float e[16];
        float Z = 0.0f;
        #pragma unroll
        for (int k = 0; k < 16; ++k) { e[k] = __expf(breg[k] - m); Z += e[k]; }
        #pragma unroll
        for (int o = 16; o > 0; o >>= 1) Z += __shfl_xor(Z, o, 32);

        const float sc = (float)Ic / Z;
        #pragma unroll
        for (int k = 0; k < 16; ++k) cI[j][v + 32 * k] = e[k] * sc;
        __syncthreads();   // make cross-lane LDS writes visible (s_wait_dscnt)

        // ---- s[j,v] = sum_i c[i,j] * uhat[b,i,j,v] ----
        float s = 0.0f;
        for (int i = 0; i < Ic; i += 4) {
            #pragma unroll
            for (int q = 0; q < 4; ++q)
                s = fmaf(cI[j][i + q], up[(i + q) * (Jc * Vc)], s);
        }

        // ---- squash along v ----
        float n2 = s * s;
        #pragma unroll
        for (int o = 16; o > 0; o >>= 1) n2 += __shfl_xor(n2, o, 32);
        vjv = s * (n2 / (1.0f + n2) * rsqrtf(n2 + 1e-9f));

        // ---- b[i,j] += <uhat[b,i,j,:], v[j,:]>  (skip on last iter) ----
        if (r < 2) {
            for (int i = 0; i < Ic; ++i) {
                float p = up[i * (Jc * Vc)] * vjv;
                #pragma unroll
                for (int o = 16; o > 0; o >>= 1) p += __shfl_xor(p, o, 32);
                if ((i & 31) == v) breg[i >> 5] += p;
            }
            __syncthreads();
        }
    }

    out[(b * Jc + j) * Vc + v] = vjv;
}

// ---------------------------------------------------------------------------
extern "C" void kernel_launch(void* const* d_in, const int* in_sizes, int n_in,
                              void* d_out, int out_size, void* d_ws, size_t ws_size,
                              hipStream_t stream)
{
    const float* x = (const float*)d_in[0];   // [128,512,256,1] fp32
    const float* W = (const float*)d_in[1];   // [1,512,32,256,32] fp32
    float* out  = (float*)d_out;              // [128,32,32,1] fp32
    float* uhat = (float*)d_ws;               // needs 128*512*32*32*4 = 256 MB scratch

    caps_uhat_gemm<<<dim3(Ic), dim3(256), 0, stream>>>(x, W, uhat);
    caps_routing<<<dim3(Bc), dim3(1024), 0, stream>>>(uhat, out);
}
